// PolynomialAttention_11166914970077
// MI455X (gfx1250) — matrix operand order repaired and checked by reading, not compiled, not run on hardware
//
#include <hip/hip_runtime.h>

typedef __attribute__((ext_vector_type(16))) __bf16 v16bf;
typedef __attribute__((ext_vector_type(8)))  float  v8f;

// ---------------- problem constants ----------------
constexpr int   B_   = 4;
constexpr int   N_   = 1024;
constexpr int   C_   = 512;
constexpr int   H_   = 8;
constexpr int   HD_  = 64;
constexpr float SCALE_  = 0.125f;   // HD^-0.5
constexpr float LN_EPS_ = 1e-5f;

constexpr int   BN_  = B_ * N_;       // 4096 rows
constexpr int   C3_  = 3 * C_;        // 1536

// ---------------- workspace layout (bytes) ----------------
constexpr size_t OFF_XB     = 0;                                  // x bf16 [4096][512]
constexpr size_t OFF_WQKVT  = OFF_XB    + (size_t)BN_ * C_ * 2;   // WqkvT bf16 [1536][512]
constexpr size_t OFF_WPROJT = OFF_WQKVT + (size_t)C3_ * C_ * 2;   // WprojT bf16 [512][512]
constexpr size_t OFF_Q      = OFF_WPROJT + (size_t)C_ * C_ * 2;   // q bf16 [B*H][N][HD]
constexpr size_t OFF_K      = OFF_Q + (size_t)B_*H_*N_*HD_*2;     // k bf16 [B*H][N][HD]
constexpr size_t OFF_VT     = OFF_K + (size_t)B_*H_*N_*HD_*2;     // vT bf16 [B*H][HD][N]
constexpr size_t OFF_O      = OFF_VT + (size_t)B_*H_*HD_*N_*2;    // O bf16 [4096][512]

// ---------------- WMMA helpers (wave32, gfx1250 layouts) ----------------
__device__ __forceinline__ v8f wmma_bf16(v16bf a, v16bf b, v8f c) {
    return __builtin_amdgcn_wmma_f32_16x16x32_bf16(
        /*neg_a=*/false, a, /*neg_b=*/false, b,
        /*c_mod=*/(short)0, c, /*reuse_a=*/false, /*reuse_b=*/false);
}

// A fragment (16x32 bf16) from row-major bf16 matrix A[row][k], leading dim lda.
// lane = half*16 + r; elements 0..7 -> K = half*8+i ; elements 8..15 -> K = 16+half*8+i
__device__ __forceinline__ v16bf load_a_bf16(const __bf16* A, int lda,
                                             int row0, int k0, int lane) {
    const int hl = lane >> 4, r = lane & 15;
    const __bf16* p = A + (size_t)(row0 + r) * lda + k0;
    v16bf a;
#pragma unroll
    for (int i = 0; i < 8; ++i) a[i]     = p[hl * 8 + i];
#pragma unroll
    for (int i = 0; i < 8; ++i) a[8 + i] = p[16 + hl * 8 + i];
    return a;
}

// A fragment from a row-major fp32 matrix (converted on the fly).
__device__ __forceinline__ v16bf load_a_f32(const float* A, int lda,
                                            int row0, int k0, int lane) {
    const int hl = lane >> 4, r = lane & 15;
    const float* p = A + (size_t)(row0 + r) * lda + k0;
    v16bf a;
#pragma unroll
    for (int i = 0; i < 8; ++i) a[i]     = (__bf16)p[hl * 8 + i];
#pragma unroll
    for (int i = 0; i < 8; ++i) a[8 + i] = (__bf16)p[16 + hl * 8 + i];
    return a;
}

// B fragment (32x16 bf16) where memory holds B^T row-major: Bt[n][k], leading dim ldb.
// lane = half*16 + n; elements 0..7 -> K = half*8+i ; elements 8..15 -> K = 16+half*8+i
__device__ __forceinline__ v16bf load_bt_bf16(const __bf16* Bt, int ldb,
                                              int n0, int k0, int lane) {
    const int hl = lane >> 4, c = lane & 15;
    const __bf16* p = Bt + (size_t)(n0 + c) * ldb + k0 + hl * 8;
    v16bf b;
#pragma unroll
    for (int i = 0; i < 8; ++i) { b[i] = p[i]; b[8 + i] = p[16 + i]; }
    return b;
}

__device__ __forceinline__ v8f zero_v8f() {
    v8f z = {0.f, 0.f, 0.f, 0.f, 0.f, 0.f, 0.f, 0.f};
    return z;
}

// ---------------- conversion kernels ----------------
__global__ void k_cvt_f32_bf16(const float* __restrict__ src,
                               __bf16* __restrict__ dst, int n) {
    int i = blockIdx.x * 256 + threadIdx.x;
    if (i < n) dst[i] = (__bf16)src[i];
}

// W[K][Nn] (row-major f32) -> Wt[Nn][K] bf16
__global__ void k_transpose_w(const float* __restrict__ W,
                              __bf16* __restrict__ Wt, int K, int Nn) {
    int i = blockIdx.x * 256 + threadIdx.x;
    if (i < K * Nn) {
        int k = i / Nn, n = i % Nn;
        Wt[(size_t)n * K + k] = (__bf16)W[i];
    }
}

// ---------------- QKV GEMM: [4096x512] x [512x1536] + bias, scatter to q/k/vT ----------------
__global__ void k_qkv(const __bf16* __restrict__ xb,
                      const __bf16* __restrict__ WqkvT,
                      const float*  __restrict__ bqkv,
                      __bf16* __restrict__ qb,
                      __bf16* __restrict__ kb,
                      __bf16* __restrict__ vtb) {
    const int tid = threadIdx.x, lane = tid & 31, wv = tid >> 5;
    const int wid = blockIdx.x * 8 + wv;
    const int NT = C3_ / 64;                 // 24 column-tiles of 64
    const int mt = wid / NT, nt = wid % NT;
    if (mt >= BN_ / 16) return;
    const int row0 = mt * 16, col0 = nt * 64;

    v8f acc[4];
#pragma unroll
    for (int t = 0; t < 4; ++t) acc[t] = zero_v8f();

    for (int k0 = 0; k0 < C_; k0 += 32) {
        v16bf a = load_a_bf16(xb, C_, row0, k0, lane);
#pragma unroll
        for (int t = 0; t < 4; ++t) {
            v16bf b = load_bt_bf16(WqkvT, C_, col0 + t * 16, k0, lane);
            acc[t] = wmma_bf16(a, b, acc[t]);
        }
    }

    const int hl = lane >> 4, cl = lane & 15;
#pragma unroll
    for (int t = 0; t < 4; ++t) {
        const int col = col0 + t * 16 + cl;
        const int t3  = col / C_;            // 0=q 1=k 2=v
        const int rem = col % C_;
        const int h   = rem / HD_;
        const int d   = rem % HD_;
        const float bias = bqkv[col];
#pragma unroll
        for (int r = 0; r < 8; ++r) {
            const int row  = row0 + r + 8 * hl;
            const int bidx = row / N_, nrow = row % N_;
            const size_t bh = (size_t)(bidx * H_ + h);
            const float val = acc[t][r] + bias;
            if (t3 == 0)      qb[(bh * N_ + nrow) * HD_ + d]  = (__bf16)val;
            else if (t3 == 1) kb[(bh * N_ + nrow) * HD_ + d]  = (__bf16)val;
            else              vtb[(bh * HD_ + d) * N_ + nrow] = (__bf16)val;
        }
    }
}

// ---------------- scores + polynomial + softmax (fused) ----------------
// grid: B*H*(N/16) blocks, 256 threads. Wave w owns columns [w*128, w*128+128).
__global__ void k_attn(const __bf16* __restrict__ qb,
                       const __bf16* __restrict__ kb,
                       const float*  __restrict__ order_w,
                       float* __restrict__ attn_out) {
    __shared__ float sred[8][16];

    const int tid = threadIdx.x, lane = tid & 31, wv = tid >> 5;
    const int bid = blockIdx.x;
    const int bh  = bid / (N_ / 16);
    const int mt  = bid % (N_ / 16);
    const int row0 = mt * 16;

    const __bf16* q  = qb + (size_t)bh * N_ * HD_;
    const __bf16* kk = kb + (size_t)bh * N_ * HD_;
    float* out = attn_out + ((size_t)bh * N_ + row0) * N_;

    // softmax over the 3 order weights (tiny, per-thread)
    const float w0 = order_w[0], w1 = order_w[1], w2 = order_w[2];
    const float mw = fmaxf(w0, fmaxf(w1, w2));
    const float e0 = __expf(w0 - mw), e1 = __expf(w1 - mw), e2 = __expf(w2 - mw);
    const float oi = 1.f / (e0 + e1 + e2);
    const float ow0 = e0 * oi, ow1 = e1 * oi, ow2 = e2 * oi;

    v8f acc[8];
#pragma unroll
    for (int t = 0; t < 8; ++t) acc[t] = zero_v8f();

    // S = Q K^T over HD=64
#pragma unroll
    for (int k0 = 0; k0 < HD_; k0 += 32) {
        v16bf a = load_a_bf16(q, HD_, row0, k0, lane);
#pragma unroll
        for (int t = 0; t < 8; ++t) {
            v16bf b = load_bt_bf16(kk, HD_, wv * 128 + t * 16, k0, lane);
            acc[t] = wmma_bf16(a, b, acc[t]);
        }
    }

    // scale + polynomial: ow0*s + ow1*s^2 + ow2*s^3
#pragma unroll
    for (int t = 0; t < 8; ++t)
#pragma unroll
        for (int r = 0; r < 8; ++r) {
            const float s = acc[t][r] * SCALE_;
            acc[t][r] = s * (ow0 + s * (ow1 + s * ow2));
        }

    const int hl = lane >> 4, cl = lane & 15;

    // ---- row max (8 rows per half-wave) ----
    float gmax[8];
#pragma unroll
    for (int r = 0; r < 8; ++r) {
        float m = acc[0][r];
#pragma unroll
        for (int t = 1; t < 8; ++t) m = fmaxf(m, acc[t][r]);
#pragma unroll
        for (int off = 1; off < 16; off <<= 1)
            m = fmaxf(m, __shfl_xor(m, off, 32));
        gmax[r] = m;
    }
    if (cl == 0) {
#pragma unroll
        for (int r = 0; r < 8; ++r) sred[wv][r + 8 * hl] = gmax[r];
    }
    __syncthreads();
#pragma unroll
    for (int r = 0; r < 8; ++r) {
        float m = sred[0][r + 8 * hl];
#pragma unroll
        for (int w = 1; w < 8; ++w) m = fmaxf(m, sred[w][r + 8 * hl]);
        gmax[r] = m;
    }
    __syncthreads();

    // ---- exp + row sum ----
    float gsum[8];
#pragma unroll
    for (int r = 0; r < 8; ++r) gsum[r] = 0.f;
#pragma unroll
    for (int t = 0; t < 8; ++t)
#pragma unroll
        for (int r = 0; r < 8; ++r) {
            const float p = __expf(acc[t][r] - gmax[r]);
            acc[t][r] = p;
            gsum[r] += p;
        }
#pragma unroll
    for (int r = 0; r < 8; ++r) {
        float s = gsum[r];
#pragma unroll
        for (int off = 1; off < 16; off <<= 1)
            s += __shfl_xor(s, off, 32);
        gsum[r] = s;
    }
    if (cl == 0) {
#pragma unroll
        for (int r = 0; r < 8; ++r) sred[wv][r + 8 * hl] = gsum[r];
    }
    __syncthreads();
#pragma unroll
    for (int r = 0; r < 8; ++r) {
        float s = 0.f;
#pragma unroll
        for (int w = 0; w < 8; ++w) s += sred[w][r + 8 * hl];
        gsum[r] = 1.f / s;
    }

    // ---- normalized probabilities -> d_out (fp32, required output) ----
#pragma unroll
    for (int t = 0; t < 8; ++t) {
        const int col = wv * 128 + t * 16 + cl;
#pragma unroll
        for (int r = 0; r < 8; ++r) {
            const int row = r + 8 * hl;
            out[(size_t)row * N_ + col] = acc[t][r] * gsum[r];
        }
    }
}

// ---------------- O = P @ V  (P read back fp32 from d_out, V pre-transposed) ----------------
__global__ void k_pv(const float* __restrict__ attn,
                     const __bf16* __restrict__ vtb,
                     __bf16* __restrict__ ob) {
    const int tid = threadIdx.x, lane = tid & 31, wv = tid >> 5;
    const int wid = blockIdx.x * 8 + wv;              // B*H*(N/16) = 2048 waves
    const int bh  = wid / (N_ / 16);
    const int mt  = wid % (N_ / 16);
    if (bh >= B_ * H_) return;
    const int row0 = mt * 16;

    const float*  P  = attn + (size_t)bh * N_ * N_;
    const __bf16* vt = vtb  + (size_t)bh * HD_ * N_;

    v8f acc[4];
#pragma unroll
    for (int t = 0; t < 4; ++t) acc[t] = zero_v8f();

    for (int k0 = 0; k0 < N_; k0 += 32) {
        v16bf a = load_a_f32(P, N_, row0, k0, lane);
#pragma unroll
        for (int t = 0; t < 4; ++t) {
            v16bf b = load_bt_bf16(vt, N_, t * 16, k0, lane);
            acc[t] = wmma_bf16(a, b, acc[t]);
        }
    }

    const int hl = lane >> 4, cl = lane & 15;
    const int bidx = bh / H_, h = bh % H_;
#pragma unroll
    for (int t = 0; t < 4; ++t) {
        const int d = t * 16 + cl;
#pragma unroll
        for (int r = 0; r < 8; ++r) {
            const int nrow = row0 + r + 8 * hl;
            ob[((size_t)(bidx * N_ + nrow)) * C_ + h * HD_ + d] = (__bf16)acc[t][r];
        }
    }
}

// ---------------- proj GEMM + bias + residual + LayerNorm (fused) ----------------
// one block per 16 output rows; 8 waves cover all 512 columns.
__global__ void k_proj_ln(const __bf16* __restrict__ ob,
                          const __bf16* __restrict__ WprojT,
                          const float*  __restrict__ bproj,
                          const float*  __restrict__ x,
                          const float*  __restrict__ gamma,
                          const float*  __restrict__ beta,
                          float* __restrict__ yout) {
    __shared__ float yt[16][C_];        // 32 KB
    __shared__ float rsum[16][16], rsq[16][16];
    __shared__ float mu_s[16], rs_s[16];

    const int tid = threadIdx.x, lane = tid & 31, wv = tid >> 5;
    const int row0 = blockIdx.x * 16;
    const int col0 = wv * 64;

    v8f acc[4];
#pragma unroll
    for (int t = 0; t < 4; ++t) acc[t] = zero_v8f();

    for (int k0 = 0; k0 < C_; k0 += 32) {
        v16bf a = load_a_bf16(ob, C_, row0, k0, lane);
#pragma unroll
        for (int t = 0; t < 4; ++t) {
            v16bf b = load_bt_bf16(WprojT, C_, col0 + t * 16, k0, lane);
            acc[t] = wmma_bf16(a, b, acc[t]);
        }
    }

    const int hl = lane >> 4, cl = lane & 15;
#pragma unroll
    for (int t = 0; t < 4; ++t) {
        const int col = col0 + t * 16 + cl;
        const float bias = bproj[col];
#pragma unroll
        for (int r = 0; r < 8; ++r) {
            const int row = r + 8 * hl;
            yt[row][col] = acc[t][r] + bias + x[(size_t)(row0 + row) * C_ + col];
        }
    }
    __syncthreads();

    // per-row mean / variance: 16 threads x 32 columns each
    {
        const int r = tid >> 4, seg = tid & 15;
        float s = 0.f, sq = 0.f;
#pragma unroll
        for (int i = 0; i < 32; ++i) {
            const float v = yt[r][seg * 32 + i];
            s += v; sq += v * v;
        }
        rsum[r][seg] = s; rsq[r][seg] = sq;
    }
    __syncthreads();
    if (tid < 16) {
        float s = 0.f, sq = 0.f;
#pragma unroll
        for (int i = 0; i < 16; ++i) { s += rsum[tid][i]; sq += rsq[tid][i]; }
        const float mu  = s / (float)C_;
        const float var = sq / (float)C_ - mu * mu;
        mu_s[tid] = mu;
        rs_s[tid] = rsqrtf(var + LN_EPS_);
    }
    __syncthreads();
    {
        const int r = tid >> 4, seg = tid & 15;
        const float mu = mu_s[r], rstd = rs_s[r];
#pragma unroll
        for (int i = 0; i < 32; ++i) {
            const int c = seg * 32 + i;
            yout[(size_t)(row0 + r) * C_ + c] =
                (yt[r][c] - mu) * rstd * gamma[c] + beta[c];
        }
    }
}

// ---------------- host launcher ----------------
extern "C" void kernel_launch(void* const* d_in, const int* in_sizes, int n_in,
                              void* d_out, int out_size, void* d_ws, size_t ws_size,
                              hipStream_t stream) {
    (void)in_sizes; (void)n_in; (void)out_size; (void)ws_size;

    const float* x     = (const float*)d_in[0];
    const float* Wqkv  = (const float*)d_in[1];
    const float* bqkv  = (const float*)d_in[2];
    const float* ordw  = (const float*)d_in[3];
    const float* Wproj = (const float*)d_in[4];
    const float* bproj = (const float*)d_in[5];
    const float* gamma = (const float*)d_in[6];
    const float* beta  = (const float*)d_in[7];

    float* yout     = (float*)d_out;                       // [B,N,C]
    float* attn_out = yout + (size_t)B_ * N_ * C_;         // [B,H,N,N]

    char* ws = (char*)d_ws;
    __bf16* xb     = (__bf16*)(ws + OFF_XB);
    __bf16* WqkvT  = (__bf16*)(ws + OFF_WQKVT);
    __bf16* WprojT = (__bf16*)(ws + OFF_WPROJT);
    __bf16* qb     = (__bf16*)(ws + OFF_Q);
    __bf16* kb     = (__bf16*)(ws + OFF_K);
    __bf16* vtb    = (__bf16*)(ws + OFF_VT);
    __bf16* ob     = (__bf16*)(ws + OFF_O);

    // 0) precision conversion / operand transposes
    {
        const int n = BN_ * C_;                            // 2,097,152
        k_cvt_f32_bf16<<<(n + 255) / 256, 256, 0, stream>>>(x, xb, n);
    }
    {
        const int n = C_ * C3_;                            // 786,432
        k_transpose_w<<<(n + 255) / 256, 256, 0, stream>>>(Wqkv, WqkvT, C_, C3_);
    }
    {
        const int n = C_ * C_;                             // 262,144
        k_transpose_w<<<(n + 255) / 256, 256, 0, stream>>>(Wproj, WprojT, C_, C_);
    }

    // 1) QKV GEMM: (BN/16)*(C3/64) = 256*24 = 6144 waves -> 768 blocks of 8 waves
    k_qkv<<<768, 256, 0, stream>>>(xb, WqkvT, bqkv, qb, kb, vtb);

    // 2) fused scores + polynomial + softmax: B*H*(N/16) = 2048 blocks
    k_attn<<<B_ * H_ * (N_ / 16), 256, 0, stream>>>(qb, kb, ordw, attn_out);

    // 3) PV: 2048 waves -> 256 blocks
    k_pv<<<256, 256, 0, stream>>>(attn_out, vtb, ob);

    // 4) proj + residual + LayerNorm: BN/16 = 256 blocks
    k_proj_ln<<<BN_ / 16, 256, 0, stream>>>(ob, WprojT, bproj, x, gamma, beta, yout);
}
